// Mono3DVGv2DecoderLayer_61426622268101
// MI455X (gfx1250) — matrix operand order repaired and checked by reading, not compiled, not run on hardware
//
#include <hip/hip_runtime.h>
#include <hip/hip_bf16.h>

// ---------------- CDNA5 WMMA plumbing (wave32, gfx1250) ----------------
typedef _Float16 v16h __attribute__((ext_vector_type(16)));
typedef _Float16 h8   __attribute__((ext_vector_type(8)));
typedef _Float16 h4   __attribute__((ext_vector_type(4)));
typedef float    v8f  __attribute__((ext_vector_type(8)));
typedef float    f4   __attribute__((ext_vector_type(4)));

#define DDIM 256
#define NHEAD 8
#define DH 32
#define BB 32
#define LQQ 512
#define LTT 128
#define LDD 6400
#define SS  8500
#define FFD 2048

__device__ __forceinline__ v8f wmma16(v16h a, v16h b, v8f c) {
  // D = A(16x32 f16) * B(32x16 f16) + C(16x16 f32)
  return __builtin_amdgcn_wmma_f32_16x16x32_f16(false, a, false, b, (short)0, c,
                                                false, false);
}

// A-matrix frag (16x32 f16). ISA 7.12.2: lane row = lane%16, K map:
// e<8 -> half*8+e ; e>=8 -> 16+half*8+(e-8). Two ds_load_b128.
__device__ __forceinline__ v16h load_a_frag(const _Float16* rowp, int half) {
  h8 lo = *(const h8*)(rowp + half * 8);
  h8 hi = *(const h8*)(rowp + 16 + half * 8);
  return __builtin_shufflevector(lo, hi, 0,1,2,3,4,5,6,7,8,9,10,11,12,13,14,15);
}

// B-matrix frag (32x16 f16): col=lane%16, K map same as A: e<8 -> half*8+e ; e>=8 -> 16+half*8+(e-8).
__device__ __forceinline__ v16h load_b_frag(const _Float16* colp, int half) {
  h8 lo = *(const h8*)(colp + half * 8);
  h8 hi = *(const h8*)(colp + 16 + half * 8);
  return __builtin_shufflevector(lo, hi, 0,1,2,3,4,5,6,7,8,9,10,11,12,13,14,15);
}

// ---------------- generic GEMM: C[M,N] = (A1 (+A2)) @ W[N,K]^T + bias ----------------
// 128 threads = 4 waves, 64x64 tile, 2x2 WMMA frags/wave, K-step 32,
// double-buffered LDS with register staging -> ONE barrier per K-step.
template <bool HAS_A2, bool RELU>
__global__ __launch_bounds__(128)
void gemm_kernel(const float* __restrict__ A1, const float* __restrict__ A2,
                 const float* __restrict__ W,  const float* __restrict__ bias,
                 float* __restrict__ C, int M, int N, int K)
{
  __shared__ _Float16 ldsA[2][64][32];
  __shared__ _Float16 ldsB[2][64][32];
  const int tid  = threadIdx.x;
  const int m0   = blockIdx.x * 64;
  const int n0   = blockIdx.y * 64;
  const int w    = tid >> 5, lane = tid & 31;
  const int half = lane >> 4, rc = lane & 15;
  const int wr   = (w >> 1) * 32, wc = (w & 1) * 32;

  const v8f vz = {0.f,0.f,0.f,0.f,0.f,0.f,0.f,0.f};
  v8f acc[2][2];
  acc[0][0]=vz; acc[0][1]=vz; acc[1][0]=vz; acc[1][1]=vz;

  // 64x32 f32 tile = 512 float4 chunks; 128 threads -> 4 chunks each.
  h4 ra[4], rb[4];
  auto stage = [&](int kt) {
#pragma unroll
    for (int j = 0; j < 4; ++j) {
      int ci = tid + j * 128;
      int r = ci >> 3, c4 = (ci & 7) * 4;
      f4 a = *(const f4*)&A1[(size_t)(m0 + r) * K + kt + c4];
      if (HAS_A2) a += *(const f4*)&A2[(size_t)(m0 + r) * K + kt + c4];
      ra[j] = __builtin_convertvector(a, h4);
      f4 b = *(const f4*)&W[(size_t)(n0 + r) * K + kt + c4];
      rb[j] = __builtin_convertvector(b, h4);
    }
  };

  stage(0);
  int p = 0;
  for (int kt = 0; kt < K; kt += 32) {
#pragma unroll
    for (int j = 0; j < 4; ++j) {
      int ci = tid + j * 128;
      int r = ci >> 3, c4 = (ci & 7) * 4;
      *(h4*)&ldsA[p][r][c4] = ra[j];
      *(h4*)&ldsB[p][r][c4] = rb[j];
    }
    __syncthreads();
    if (kt + 32 < K) stage(kt + 32);   // HBM loads overlap WMMA below

    v16h af0 = load_a_frag(&ldsA[p][wr + rc][0], half);
    v16h af1 = load_a_frag(&ldsA[p][wr + 16 + rc][0], half);
    v16h bf0 = load_b_frag(&ldsB[p][wc + rc][0], half);
    v16h bf1 = load_b_frag(&ldsB[p][wc + 16 + rc][0], half);
    acc[0][0] = wmma16(af0, bf0, acc[0][0]);
    acc[0][1] = wmma16(af0, bf1, acc[0][1]);
    acc[1][0] = wmma16(af1, bf0, acc[1][0]);
    acc[1][1] = wmma16(af1, bf1, acc[1][1]);
    p ^= 1;
  }

#pragma unroll
  for (int j = 0; j < 2; ++j) {
    int n = n0 + wc + j * 16 + rc;
    float bv = bias[n];
#pragma unroll
    for (int i = 0; i < 2; ++i)
#pragma unroll
      for (int r = 0; r < 8; ++r) {
        float v = acc[i][j][r] + bv;
        if (RELU) v = fmaxf(v, 0.f);
        C[(size_t)(m0 + wr + i * 16 + half * 8 + r) * N + n] = v;
      }
  }
}

// ---------------- flash attention: Lq=512 fixed, dh=32, runtime Lk + optional mask ------
// 4 waves/block share one (b,h); each wave owns a 16-row q tile; 32 keys/iter;
// double-buffered K/V tiles (one barrier per iter) + global_prefetch of the block after next.
__global__ __launch_bounds__(128)
void attn_kernel(const float* __restrict__ Q, const float* __restrict__ Kp,
                 const float* __restrict__ Vp, const unsigned char* __restrict__ mask,
                 float* __restrict__ O, int Lk)
{
  __shared__ _Float16 ldsK[2][32][32];   // [key][dh]
  __shared__ _Float16 ldsVt[2][32][32];  // [dh][key]
  __shared__ _Float16 ldsP[4][16][32];
  const int tid = threadIdx.x, w = tid >> 5, lane = tid & 31;
  const int half = lane >> 4, rc = lane & 15;
  const int QG = (LQQ / 16) / 4;
  const int bid = blockIdx.x;
  const int b = bid / (NHEAD * QG);
  const int rem = bid % (NHEAD * QG);
  const int h = rem / QG;
  const int qt = (rem % QG) * 4 + w;
  const int qrow0 = b * LQQ + qt * 16;

  // Q A-frag, pre-scaled by 1/sqrt(dh)
  const float scale = 0.17677669529663687f;
  v16h qf;
  {
    const float* qp = &Q[(size_t)(qrow0 + rc) * DDIM + h * DH];
    f4 q0 = *(const f4*)(qp + half * 8);
    f4 q1 = *(const f4*)(qp + half * 8 + 4);
    f4 q2 = *(const f4*)(qp + 16 + half * 8);
    f4 q3 = *(const f4*)(qp + 16 + half * 8 + 4);
#pragma unroll
    for (int e = 0; e < 4; ++e) {
      qf[e]      = (_Float16)(q0[e] * scale);
      qf[4 + e]  = (_Float16)(q1[e] * scale);
      qf[8 + e]  = (_Float16)(q2[e] * scale);
      qf[12 + e] = (_Float16)(q3[e] * scale);
    }
  }

  float m[8], l[8];
  const v8f vz = {0.f,0.f,0.f,0.f,0.f,0.f,0.f,0.f};
  v8f o0 = vz, o1 = vz;
#pragma unroll
  for (int r = 0; r < 8; ++r) { m[r] = -1e30f; l[r] = 0.f; }

  // K/V tile staging: 32x32 f32 = 256 float4 chunks; 128 threads -> 2 chunks each.
  h4 rk[2], rv[2];
  auto stage_kv = [&](int kb) {
#pragma unroll
    for (int j = 0; j < 2; ++j) {
      int ci = tid + j * 128;
      int kk = ci >> 3, c4 = (ci & 7) * 4;
      size_t gi = (size_t)(b * Lk + kb + kk) * DDIM + h * DH + c4;
      rk[j] = __builtin_convertvector(*(const f4*)&Kp[gi], h4);
      rv[j] = __builtin_convertvector(*(const f4*)&Vp[gi], h4);
    }
  };

  stage_kv(0);
  int p = 0;
  for (int kb = 0; kb < Lk; kb += 32) {
#pragma unroll
    for (int j = 0; j < 2; ++j) {
      int ci = tid + j * 128;
      int kk = ci >> 3, c4 = (ci & 7) * 4;
      *(h4*)&ldsK[p][kk][c4] = rk[j];
#pragma unroll
      for (int u = 0; u < 4; ++u) ldsVt[p][c4 + u][kk] = rv[j][u];
    }
    __syncthreads();
    if (kb + 32 < Lk) {
      stage_kv(kb + 32);
      if (kb + 64 < Lk) {  // prefetch the block after next into cache
        size_t gp = (size_t)(b * Lk + kb + 64 + (tid >> 3)) * DDIM + h * DH + (tid & 7) * 4;
        __builtin_prefetch(&Kp[gp], 0, 0);
        __builtin_prefetch(&Vp[gp], 0, 0);
      }
    }

    v16h bk0 = load_b_frag(&ldsK[p][rc][0], half);
    v16h bk1 = load_b_frag(&ldsK[p][16 + rc][0], half);
    v8f s0 = wmma16(qf, bk0, vz);
    v8f s1 = wmma16(qf, bk1, vz);

    if (mask) {
      if (mask[b * Lk + kb + rc]) {
#pragma unroll
        for (int r = 0; r < 8; ++r) s0[r] = -1e9f;
      }
      if (mask[b * Lk + kb + 16 + rc]) {
#pragma unroll
        for (int r = 0; r < 8; ++r) s1[r] = -1e9f;
      }
    }

    float mn[8];
#pragma unroll
    for (int r = 0; r < 8; ++r) mn[r] = fmaxf(s0[r], s1[r]);
#pragma unroll
    for (int d = 1; d < 16; d <<= 1)
#pragma unroll
      for (int r = 0; r < 8; ++r) mn[r] = fmaxf(mn[r], __shfl_xor(mn[r], d, 32));

    float p0[8], p1[8], ls[8];
#pragma unroll
    for (int r = 0; r < 8; ++r) {
      float mx = fmaxf(m[r], mn[r]);
      float corr = __expf(m[r] - mx);
      m[r] = mx;
      p0[r] = __expf(s0[r] - mx);
      p1[r] = __expf(s1[r] - mx);
      ls[r] = p0[r] + p1[r];
      l[r] *= corr;
      o0[r] *= corr;
      o1[r] *= corr;
    }
#pragma unroll
    for (int d = 1; d < 16; d <<= 1)
#pragma unroll
      for (int r = 0; r < 8; ++r) ls[r] += __shfl_xor(ls[r], d, 32);
#pragma unroll
    for (int r = 0; r < 8; ++r) l[r] += ls[r];

    // P: D-layout -> A-layout via per-wave private LDS tile (same-wave DS order)
#pragma unroll
    for (int r = 0; r < 8; ++r) {
      ldsP[w][half * 8 + r][rc] = (_Float16)p0[r];
      ldsP[w][half * 8 + r][16 + rc] = (_Float16)p1[r];
    }
    v16h pf  = load_a_frag(&ldsP[w][rc][0], half);
    v16h bv0 = load_b_frag(&ldsVt[p][rc][0], half);
    v16h bv1 = load_b_frag(&ldsVt[p][16 + rc][0], half);
    o0 = wmma16(pf, bv0, o0);
    o1 = wmma16(pf, bv1, o1);
    p ^= 1;
  }

#pragma unroll
  for (int r = 0; r < 8; ++r) {
    float inv = 1.f / l[r];
    size_t mrow = (size_t)(qrow0 + half * 8 + r) * DDIM + h * DH;
    O[mrow + rc] = o0[r] * inv;
    O[mrow + 16 + rc] = o1[r] * inv;
  }
}

// ---------------- LayerNorm(x + res) * g + b ; wave-per-row, b128 IO ----------------
__global__ __launch_bounds__(128)
void ln_kernel(const float* __restrict__ X, const float* __restrict__ R,
               const float* __restrict__ g, const float* __restrict__ be,
               float* __restrict__ Out, int Mrows)
{
  int w = threadIdx.x >> 5, lane = threadIdx.x & 31;
  int row = blockIdx.x * 4 + w;
  if (row >= Mrows) return;
  const float* xp = X + (size_t)row * DDIM + lane * 8;
  const float* rp = R + (size_t)row * DDIM + lane * 8;
  f4 v0 = *(const f4*)xp + *(const f4*)rp;
  f4 v1 = *(const f4*)(xp + 4) + *(const f4*)(rp + 4);
  float s = v0[0]+v0[1]+v0[2]+v0[3]+v1[0]+v1[1]+v1[2]+v1[3];
#pragma unroll
  for (int d = 1; d < 32; d <<= 1) s += __shfl_xor(s, d, 32);
  float mean = s * (1.f / 256.f);
  float vs = 0.f;
#pragma unroll
  for (int i = 0; i < 4; ++i) { float t0 = v0[i]-mean, t1 = v1[i]-mean; vs += t0*t0 + t1*t1; }
#pragma unroll
  for (int d = 1; d < 32; d <<= 1) vs += __shfl_xor(vs, d, 32);
  float inv = rsqrtf(vs * (1.f / 256.f) + 1e-5f);
  f4 g0 = *(const f4*)(g + lane * 8),  g1 = *(const f4*)(g + lane * 8 + 4);
  f4 b0 = *(const f4*)(be + lane * 8), b1 = *(const f4*)(be + lane * 8 + 4);
  f4 o0, o1;
#pragma unroll
  for (int i = 0; i < 4; ++i) {
    o0[i] = (v0[i] - mean) * inv * g0[i] + b0[i];
    o1[i] = (v1[i] - mean) * inv * g1[i] + b1[i];
  }
  float* op = Out + (size_t)row * DDIM + lane * 8;
  *(f4*)op = o0;
  *(f4*)(op + 4) = o1;
}

// ---------------- ms-deform bilinear sampling: wave per (b,q,head), lane = channel ------
__global__ __launch_bounds__(256)
void msdeform_kernel(const float* __restrict__ VMS, const float* __restrict__ OFF,
                     const float* __restrict__ AW, const float* __restrict__ REF,
                     float* __restrict__ Out)
{
  const int Wl[4] = {80, 40, 20, 10};
  const int Hl[4] = {80, 40, 20, 10};
  const int st[4] = {0, 6400, 8000, 8400};
  int w = threadIdx.x >> 5, lane = threadIdx.x & 31;
  int item = blockIdx.x * 8 + w;          // over (b*LQ + q)*NH + h
  int h = item & (NHEAD - 1);
  int qrow = item >> 3;
  int b = qrow >> 9;                      // / LQQ

  const float* awp = AW + (size_t)qrow * (NHEAD * 16) + h * 16;
  float wts[16];
  float mx = -1e30f;
#pragma unroll
  for (int i = 0; i < 16; ++i) { wts[i] = awp[i]; mx = fmaxf(mx, wts[i]); }
  float sum = 0.f;
#pragma unroll
  for (int i = 0; i < 16; ++i) { wts[i] = __expf(wts[i] - mx); sum += wts[i]; }
  float invs = 1.f / sum;

  const float* offp = OFF + (size_t)qrow * DDIM + h * 32;
  float acc = 0.f;
#pragma unroll
  for (int lev = 0; lev < 4; ++lev) {
    float refx = REF[((size_t)qrow * 4 + lev) * 2 + 0];
    float refy = REF[((size_t)qrow * 4 + lev) * 2 + 1];
    float fW = (float)Wl[lev], fH = (float)Hl[lev];
#pragma unroll
    for (int pp = 0; pp < 4; ++pp) {
      float ox = offp[(lev * 4 + pp) * 2 + 0];
      float oy = offp[(lev * 4 + pp) * 2 + 1];
      float lx = (refx + ox / fW) * fW - 0.5f;
      float ly = (refy + oy / fH) * fH - 0.5f;
      float x0 = floorf(lx), y0 = floorf(ly);
      float fx = lx - x0, fy = ly - y0;
      int x0i = (int)x0, y0i = (int)y0;
      float aw = wts[lev * 4 + pp] * invs;
      float cw[4] = {(1.f - fx) * (1.f - fy), fx * (1.f - fy),
                     (1.f - fx) * fy,          fx * fy};
      const int dxs[4] = {0, 1, 0, 1};
      const int dys[4] = {0, 0, 1, 1};
#pragma unroll
      for (int c = 0; c < 4; ++c) {
        int xc = x0i + dxs[c], yc = y0i + dys[c];
        float valid = (xc >= 0 && xc < Wl[lev] && yc >= 0 && yc < Hl[lev]) ? 1.f : 0.f;
        int xcc = min(max(xc, 0), Wl[lev] - 1);
        int ycc = min(max(yc, 0), Hl[lev] - 1);
        int idx = ycc * Wl[lev] + xcc;
        float val = VMS[(size_t)(b * SS + st[lev] + idx) * DDIM + h * 32 + lane];
        acc += cw[c] * valid * aw * val;
      }
    }
  }
  Out[(size_t)qrow * DDIM + h * 32 + lane] = acc;
}

// ---------------------------------- host orchestration ----------------------------------
static inline void launch_gemm(const float* A1, const float* A2, const float* W,
                               const float* bias, float* C, int M, int N, int K,
                               int relu, hipStream_t s) {
  dim3 grid(M / 64, N / 64);
  if (A2)       gemm_kernel<true,  false><<<grid, 128, 0, s>>>(A1, A2, W, bias, C, M, N, K);
  else if (relu) gemm_kernel<false, true><<<grid, 128, 0, s>>>(A1, A2, W, bias, C, M, N, K);
  else          gemm_kernel<false, false><<<grid, 128, 0, s>>>(A1, A2, W, bias, C, M, N, K);
}

extern "C" void kernel_launch(void* const* d_in, const int* in_sizes, int n_in,
                              void* d_out, int out_size, void* d_ws, size_t ws_size,
                              hipStream_t stream) {
  const float* x_h    = (const float*)d_in[0];
  const float* x_pos  = (const float*)d_in[1];
  const float* x_text = (const float*)d_in[2];
  const unsigned char* m_text  = (const unsigned char*)d_in[3];
  const float* x_de   = (const float*)d_in[4];
  const unsigned char* m_depth = (const unsigned char*)d_in[5];
  const float* x_dak  = (const float*)d_in[6];
  const float* x_enc  = (const float*)d_in[7];
  const float* x_ref  = (const float*)d_in[8];
  // d_in[9] spatial_shapes, d_in[10] level_start_index: compile-time constants here.
  const float* depth_in_w  = (const float*)d_in[11];
  const float* depth_in_b  = (const float*)d_in[12];
  const float* depth_out_w = (const float*)d_in[13];
  const float* depth_out_b = (const float*)d_in[14];
  const float* text_in_w   = (const float*)d_in[15];
  const float* text_in_b   = (const float*)d_in[16];
  const float* text_out_w  = (const float*)d_in[17];
  const float* text_out_b  = (const float*)d_in[18];
  const float* self_in_w   = (const float*)d_in[19];
  const float* self_in_b   = (const float*)d_in[20];
  const float* self_out_w  = (const float*)d_in[21];
  const float* self_out_b  = (const float*)d_in[22];
  const float* off_w = (const float*)d_in[23];
  const float* off_b = (const float*)d_in[24];
  const float* aw_w  = (const float*)d_in[25];
  const float* aw_b  = (const float*)d_in[26];
  const float* val_w = (const float*)d_in[27];
  const float* val_b = (const float*)d_in[28];
  const float* out_w = (const float*)d_in[29];
  const float* out_b = (const float*)d_in[30];
  const float* fc1_w = (const float*)d_in[31];
  const float* fc1_b = (const float*)d_in[32];
  const float* fc2_w = (const float*)d_in[33];
  const float* fc2_b = (const float*)d_in[34];
  const float* ln_g[5] = {(const float*)d_in[35], (const float*)d_in[37],
                          (const float*)d_in[39], (const float*)d_in[41],
                          (const float*)d_in[43]};
  const float* ln_b[5] = {(const float*)d_in[36], (const float*)d_in[38],
                          (const float*)d_in[40], (const float*)d_in[42],
                          (const float*)d_in[44]};

  const int MQ = BB * LQQ;   // 16384
  const int MT = BB * LTT;   // 4096
  const int MD = BB * LDD;   // 204800
  const int MS = BB * SS;    // 272000

  float* ws = (float*)d_ws;
  size_t o = 0;
  float* hbuf  = ws + o; o += (size_t)MQ * DDIM;     // current hidden
  float* xbuf  = ws + o; o += (size_t)MQ * DDIM;     // block output pre-LN
  float* qbuf  = ws + o; o += (size_t)MQ * DDIM;     // Q projection
  float* aobuf = ws + o; o += (size_t)MQ * DDIM;     // attention output pre-proj
  float* offbf = ws + o; o += (size_t)MQ * DDIM;     // deform offsets (256/tok)
  float* awbf  = ws + o; o += (size_t)MQ * 128;      // deform aw logits
  float* kbuf  = ws + o; o += (size_t)MD * DDIM;     // K proj (big arena)
  float* vbuf  = ws + o; o += (size_t)MD * DDIM;     // V proj (big arena)
  float* vms   = kbuf;                               // MS*256 fits in K+V arena
  float* ffmid = kbuf;                               // MQ*FF fits in K arena

  const int QGRID = BB * NHEAD * (LQQ / 16) / 4;     // attn blocks = 2048
  const int LNGRID = MQ / 4;

  // ---- 1) depth cross-attention ----
  launch_gemm(x_h, nullptr, depth_in_w,              depth_in_b,       qbuf, MQ, DDIM, DDIM, 0, stream);
  launch_gemm(x_dak, nullptr, depth_in_w + 256*256,  depth_in_b + 256, kbuf, MD, DDIM, DDIM, 0, stream);
  launch_gemm(x_de,  nullptr, depth_in_w + 512*256,  depth_in_b + 512, vbuf, MD, DDIM, DDIM, 0, stream);
  attn_kernel<<<QGRID, 128, 0, stream>>>(qbuf, kbuf, vbuf, m_depth, aobuf, LDD);
  launch_gemm(aobuf, nullptr, depth_out_w, depth_out_b, xbuf, MQ, DDIM, DDIM, 0, stream);
  ln_kernel<<<LNGRID, 128, 0, stream>>>(xbuf, x_h, ln_g[0], ln_b[0], hbuf, MQ);

  // ---- 2) text cross-attention (q = h + pos) ----
  launch_gemm(hbuf, x_pos,  text_in_w,            text_in_b,       qbuf, MQ, DDIM, DDIM, 0, stream);
  launch_gemm(x_text, nullptr, text_in_w + 256*256, text_in_b + 256, kbuf, MT, DDIM, DDIM, 0, stream);
  launch_gemm(x_text, nullptr, text_in_w + 512*256, text_in_b + 512, vbuf, MT, DDIM, DDIM, 0, stream);
  attn_kernel<<<QGRID, 128, 0, stream>>>(qbuf, kbuf, vbuf, m_text, aobuf, LTT);
  launch_gemm(aobuf, nullptr, text_out_w, text_out_b, xbuf, MQ, DDIM, DDIM, 0, stream);
  ln_kernel<<<LNGRID, 128, 0, stream>>>(xbuf, hbuf, ln_g[1], ln_b[1], hbuf, MQ);

  // ---- 3) self-attention (q = k = h + pos, v = h) ----
  launch_gemm(hbuf, x_pos, self_in_w,            self_in_b,       qbuf, MQ, DDIM, DDIM, 0, stream);
  launch_gemm(hbuf, x_pos, self_in_w + 256*256,  self_in_b + 256, kbuf, MQ, DDIM, DDIM, 0, stream);
  launch_gemm(hbuf, nullptr, self_in_w + 512*256, self_in_b + 512, vbuf, MQ, DDIM, DDIM, 0, stream);
  attn_kernel<<<QGRID, 128, 0, stream>>>(qbuf, kbuf, vbuf, nullptr, aobuf, LQQ);
  launch_gemm(aobuf, nullptr, self_out_w, self_out_b, xbuf, MQ, DDIM, DDIM, 0, stream);
  ln_kernel<<<LNGRID, 128, 0, stream>>>(xbuf, hbuf, ln_g[2], ln_b[2], hbuf, MQ);

  // ---- 4) multi-scale deformable attention ----
  launch_gemm(x_enc, nullptr, val_w, val_b, vms, MS, DDIM, DDIM, 0, stream);
  launch_gemm(hbuf, x_pos, off_w, off_b, offbf, MQ, DDIM, DDIM, 0, stream);
  launch_gemm(hbuf, x_pos, aw_w,  aw_b,  awbf,  MQ, 128,  DDIM, 0, stream);
  msdeform_kernel<<<(BB * LQQ * NHEAD) / 8, 256, 0, stream>>>(vms, offbf, awbf, x_ref, aobuf);
  launch_gemm(aobuf, nullptr, out_w, out_b, xbuf, MQ, DDIM, DDIM, 0, stream);
  ln_kernel<<<LNGRID, 128, 0, stream>>>(xbuf, hbuf, ln_g[3], ln_b[3], hbuf, MQ);

  // ---- 5) FFN ----
  launch_gemm(hbuf, nullptr, fc1_w, fc1_b, ffmid, MQ, FFD, DDIM, 1, stream);
  launch_gemm(ffmid, nullptr, fc2_w, fc2_b, xbuf, MQ, DDIM, FFD, 0, stream);
  ln_kernel<<<LNGRID, 128, 0, stream>>>(xbuf, hbuf, ln_g[4], ln_b[4], (float*)d_out, MQ);
}
